// GNN_89395449299080
// MI455X (gfx1250) — matrix operand order repaired and checked by reading, not compiled, not run on hardware
//
#include <hip/hip_runtime.h>
#include <hip/hip_bf16.h>

typedef __attribute__((ext_vector_type(16))) _Float16 v16h;
typedef __attribute__((ext_vector_type(8)))  float    v8f;

#define NODE_D 64

// ---------------------------------------------------------------------------
// Pack a [K,64] f32 weight (optionally minus a second weight) into the exact
// per-lane WMMA B-matrix (32x16 f16 per k-step) layout:
//   dst[((s*4 + ntile)*32 + lane)*16 + idx] = W[s*32 + (idx<8 ? half*8+idx : 8+half*8+idx)][ntile*16 + (lane&15)]
// Rows k >= K are zero-padded (used for the K=16 encoder GEMM).
// ---------------------------------------------------------------------------
__global__ __launch_bounds__(256)
void pack_w_kernel(const float* __restrict__ A, const float* __restrict__ B,
                   _Float16* __restrict__ dst, int K, int ksteps) {
    int tid = blockIdx.x * blockDim.x + threadIdx.x;
    int total = ksteps * 4 * 32;
    if (tid >= total) return;
    int lane = tid & 31;
    int t    = (tid >> 5) & 3;
    int s    = tid >> 7;
    int half = lane >> 4;
    int n    = t * 16 + (lane & 15);
    _Float16* p = dst + ((size_t)(s * 4 + t) * 32 + lane) * 16;
#pragma unroll
    for (int idx = 0; idx < 16; ++idx) {
        int k = s * 32 + (idx < 8 ? half * 8 + idx : 16 + half * 8 + (idx - 8));
        float v = 0.0f;
        if (k < K) {
            v = A[k * NODE_D + n];
            if (B) v -= B[k * NODE_D + n];
        }
        p[idx] = (_Float16)v;
    }
}

// ---------------------------------------------------------------------------
// Dual-output node GEMM: Y0 = act0(X @ W0 + b0a + b0b), Y1 = act1(X @ W1 + ...)
// X is [M, KDIM] f32 row-major, KDIM in {16, 64} (compile-time); Y is [M,64].
// One wave computes a 16(M) x 64(N) tile: 4 WMMA accumulators per output.
// A tile is loaded with unconditional float4 (b128) loads and converted
// f32->f16 in registers in the ISA 16x32 A layout (lane half holds
// K = s*32 + {half*8 + j, 16 + half*8 + j}); for KDIM=16 the upper K half is
// compile-time zero.
// ---------------------------------------------------------------------------
template <int KDIM, bool DUAL>
__global__ __launch_bounds__(256)
void gemm_kernel(const float* __restrict__ X, int M,
                 const _Float16* __restrict__ Wp0, const float* __restrict__ b0a,
                 const float* __restrict__ b0b, float* __restrict__ Y0, int act0,
                 const _Float16* __restrict__ Wp1, const float* __restrict__ b1a,
                 const float* __restrict__ b1b, float* __restrict__ Y1, int act1) {
    constexpr int KSTEPS = (KDIM + 31) / 32;
    int tid  = threadIdx.x;
    int wave = tid >> 5;
    int lane = tid & 31;
    int half = lane >> 4;
    int mrow = lane & 15;
    int n0   = lane & 15;
    long tileRow = ((long)blockIdx.x * 8 + wave) * 16;
    if (tileRow >= M) return;

    v8f c0[4], c1[4];
#pragma unroll
    for (int t = 0; t < 4; ++t) {
        int n = t * 16 + n0;
        float bv0 = (b0a ? b0a[n] : 0.0f) + (b0b ? b0b[n] : 0.0f);
        float bv1 = DUAL ? ((b1a ? b1a[n] : 0.0f) + (b1b ? b1b[n] : 0.0f)) : 0.0f;
#pragma unroll
        for (int r = 0; r < 8; ++r) { c0[t][r] = bv0; c1[t][r] = bv1; }
    }

#pragma unroll
    for (int s = 0; s < KSTEPS; ++s) {
        // ---- A tile: two 8-float contiguous runs per lane, b128 loads ----
        v16h a;
        const float* base = X + (tileRow + mrow) * (long)KDIM + s * 32 + half * 8;
        {
            float4 lo0 = *(const float4*)(base + 0);
            float4 lo1 = *(const float4*)(base + 4);
#pragma unroll
            for (int j = 0; j < 4; ++j) {
                a[j]     = (_Float16)(&lo0.x)[j];
                a[4 + j] = (_Float16)(&lo1.x)[j];
            }
            if constexpr (KDIM >= 32) {
                float4 hi0 = *(const float4*)(base + 16);
                float4 hi1 = *(const float4*)(base + 20);
#pragma unroll
                for (int j = 0; j < 4; ++j) {
                    a[8 + j]  = (_Float16)(&hi0.x)[j];
                    a[12 + j] = (_Float16)(&hi1.x)[j];
                }
            } else {
#pragma unroll
                for (int j = 0; j < 8; ++j) a[8 + j] = (_Float16)0.0f;
            }
        }
        // ---- B tiles: load all 4 (or 8) first so loads clause/overlap ----
        v16h b0[4];
#pragma unroll
        for (int t = 0; t < 4; ++t)
            b0[t] = *(const v16h*)(Wp0 + ((size_t)(s * 4 + t) * 32 + lane) * 16);
        v16h b1[4];
        if constexpr (DUAL) {
#pragma unroll
            for (int t = 0; t < 4; ++t)
                b1[t] = *(const v16h*)(Wp1 + ((size_t)(s * 4 + t) * 32 + lane) * 16);
        }
#pragma unroll
        for (int t = 0; t < 4; ++t)
            c0[t] = __builtin_amdgcn_wmma_f32_16x16x32_f16(
                false, a, false, b0[t], (short)0, c0[t], false, false);
        if constexpr (DUAL) {
#pragma unroll
            for (int t = 0; t < 4; ++t)
                c1[t] = __builtin_amdgcn_wmma_f32_16x16x32_f16(
                    false, a, false, b1[t], (short)0, c1[t], false, false);
        }
    }

    // D layout: c[r] -> row = tileRow + r + 8*half, col = t*16 + n0
#pragma unroll
    for (int t = 0; t < 4; ++t) {
        int n = t * 16 + n0;
#pragma unroll
        for (int r = 0; r < 8; ++r) {
            long row = tileRow + r + 8 * half;
            float v = c0[t][r];
            if (act0) v = fmaxf(v, 0.0f);
            Y0[row * NODE_D + n] = v;
        }
    }
    if constexpr (DUAL) {
#pragma unroll
        for (int t = 0; t < 4; ++t) {
            int n = t * 16 + n0;
#pragma unroll
            for (int r = 0; r < 8; ++r) {
                long row = tileRow + r + 8 * half;
                float v = c1[t][r];
                if (act1) v = fmaxf(v, 0.0f);
                Y1[row * NODE_D + n] = v;
            }
        }
    }
}

// ---------------------------------------------------------------------------
// Fill a u32 buffer with -inf bit pattern, vectorized x4 (n must be %4 == 0).
// ---------------------------------------------------------------------------
__global__ __launch_bounds__(256)
void fill_u32_kernel(uint4* __restrict__ p, unsigned val, long n4) {
    long i = (long)blockIdx.x * blockDim.x + threadIdx.x;
    if (i < n4) p[i] = make_uint4(val, val, val, val);
}

// ---------------------------------------------------------------------------
// Edge scatter-max: one wave per edge. m[dst] = max(m[dst], a[src]) per column.
// Float max via order-preserving integer atomics (deterministic, native
// global_atomic_max_i32 / global_atomic_min_u32). Both a and m are L2-resident
// (25.6 MB each << 192 MB L2), so this runs at L2 bandwidth.
// ---------------------------------------------------------------------------
__global__ __launch_bounds__(256)
void edge_scatter_kernel(const float* __restrict__ a, const int* __restrict__ src,
                         const int* __restrict__ dst, unsigned* __restrict__ m,
                         int nEdges) {
    long gid = (long)blockIdx.x * blockDim.x + threadIdx.x;
    long e   = gid >> 5;
    int lane = (int)(gid & 31);
    if (e >= nEdges) return;
    int s = src[e];
    int d = dst[e];
    const float* ar = a + (long)s * NODE_D;
    unsigned*    mr = m + (long)d * NODE_D;
#pragma unroll
    for (int h = 0; h < 2; ++h) {
        int c = lane + h * 32;
        float v = ar[c];
        if (v >= 0.0f) atomicMax((int*)(mr + c), __float_as_int(v));
        else           atomicMin(mr + c, __float_as_uint(v));
    }
}

// ---------------------------------------------------------------------------
// x = relu( (m == -inf) ? 0 : bits(m) + b ), vectorized x4. Safe in place
// (m aliases x; elementwise same-index read/write).
// ---------------------------------------------------------------------------
__global__ __launch_bounds__(256)
void combine_kernel(const uint4* __restrict__ m, const float4* __restrict__ b,
                    float4* __restrict__ x, long n4) {
    long i = (long)blockIdx.x * blockDim.x + threadIdx.x;
    if (i >= n4) return;
    uint4  mb = m[i];
    float4 bb = b[i];
    float4 o;
    const unsigned* mp = &mb.x;
    const float*    bp = &bb.x;
    float*          op = &o.x;
#pragma unroll
    for (int j = 0; j < 4; ++j) {
        unsigned bits = mp[j];
        float v = (bits == 0xFF800000u) ? 0.0f : (__uint_as_float(bits) + bp[j]);
        op[j] = fmaxf(v, 0.0f);
    }
    x[i] = o;
}

// ---------------------------------------------------------------------------
// Final: out[i] = thresh( y1[i,:] . w2 + b2 + feats[i,15] ). One wave/node.
// ---------------------------------------------------------------------------
__global__ __launch_bounds__(256)
void decoder2_kernel(const float* __restrict__ y1, const float* __restrict__ w2,
                     const float* __restrict__ b2, const float* __restrict__ feats,
                     float* __restrict__ out, int nNodes) {
    long gid = (long)blockIdx.x * blockDim.x + threadIdx.x;
    long node = gid >> 5;
    int lane  = (int)(gid & 31);
    if (node >= nNodes) return;
    const float* r = y1 + node * NODE_D;
    float p = r[lane] * w2[lane] + r[lane + 32] * w2[lane + 32];
#pragma unroll
    for (int off = 16; off > 0; off >>= 1) p += __shfl_xor(p, off, 32);
    if (lane == 0) {
        float v = p + b2[0] + feats[node * 16 + 15];
        out[node] = (v < 0.01f) ? 0.0f : v;
    }
}

// ---------------------------------------------------------------------------
extern "C" void kernel_launch(void* const* d_in, const int* in_sizes, int n_in,
                              void* d_out, int out_size, void* d_ws, size_t ws_size,
                              hipStream_t stream) {
    const float* feats  = (const float*)d_in[0];
    const int*   src    = (const int*)d_in[1];
    const int*   dst    = (const int*)d_in[2];
    const float* enc1W  = (const float*)d_in[3];
    const float* enc1b  = (const float*)d_in[4];
    const float* enc2W  = (const float*)d_in[5];
    const float* enc2b  = (const float*)d_in[6];
    const float* thetaW = (const float*)d_in[7];
    const float* thetab = (const float*)d_in[8];
    const float* phiW   = (const float*)d_in[9];
    const float* phib   = (const float*)d_in[10];
    const float* dec1W  = (const float*)d_in[11];
    const float* dec1b  = (const float*)d_in[12];
    const float* dec2W  = (const float*)d_in[13];
    const float* dec2b  = (const float*)d_in[14];

    const int nNodes = in_sizes[0] / 16;
    const int nEdges = in_sizes[1];
    const long nodeF = (long)nNodes * NODE_D;

    // Workspace: a | b | x (x doubles as the -inf/max accumulator m) | packed W
    float* abuf = (float*)d_ws;
    float* bbuf = abuf + nodeF;
    float* xbuf = bbuf + nodeF;
    _Float16* packs = (_Float16*)(xbuf + nodeF);
    auto PK = [&](int i) { return packs + (size_t)i * 4096; };

    // ---- pack all weights into WMMA B layout (tiny, runs once per call) ----
    pack_w_kernel<<<1, 256, 0, stream>>>(enc1W, nullptr, PK(0), 16, 1);
    pack_w_kernel<<<1, 256, 0, stream>>>(enc2W, nullptr, PK(1), 64, 2);
    for (int l = 0; l < 3; ++l) {
        pack_w_kernel<<<1, 256, 0, stream>>>(thetaW + (size_t)l * 4096, nullptr,
                                             PK(2 + 2 * l), 64, 2);
        pack_w_kernel<<<1, 256, 0, stream>>>(phiW + (size_t)l * 4096,
                                             thetaW + (size_t)l * 4096,
                                             PK(3 + 2 * l), 64, 2);
    }
    pack_w_kernel<<<1, 256, 0, stream>>>(dec1W, nullptr, PK(8), 64, 2);

    const int tiles   = (nNodes + 15) / 16;
    const int gblocks = (tiles + 7) / 8;
    const long n4     = nodeF / 4;
    const int eblocks = (int)((n4 + 255) / 256);
    const int sblocks = (int)(((long)nEdges * 32 + 255) / 256);
    const int fblocks = (int)(((long)nNodes * 32 + 255) / 256);

    // ---- encoder ----
    gemm_kernel<16, false><<<gblocks, 256, 0, stream>>>(
        feats, nNodes, PK(0), enc1b, nullptr, abuf, 1,
        nullptr, nullptr, nullptr, nullptr, 0);
    gemm_kernel<64, false><<<gblocks, 256, 0, stream>>>(
        abuf, nNodes, PK(1), enc2b, nullptr, xbuf, 1,
        nullptr, nullptr, nullptr, nullptr, 0);

    // ---- 3 edge-conv layers: a = x@tW ; b = x@(pW-tW) + tb + pb ;
    //      m[dst] = max a[src] ; x = relu(m == -inf ? 0 : m + b)
    for (int l = 0; l < 3; ++l) {
        gemm_kernel<64, true><<<gblocks, 256, 0, stream>>>(
            xbuf, nNodes,
            PK(2 + 2 * l), nullptr, nullptr, abuf, 0,
            PK(3 + 2 * l), thetab + (size_t)l * 64, phib + (size_t)l * 64, bbuf, 0);
        fill_u32_kernel<<<eblocks, 256, 0, stream>>>((uint4*)xbuf, 0xFF800000u, n4);
        edge_scatter_kernel<<<sblocks, 256, 0, stream>>>(abuf, src, dst,
                                                         (unsigned*)xbuf, nEdges);
        combine_kernel<<<eblocks, 256, 0, stream>>>((const uint4*)xbuf,
                                                    (const float4*)bbuf,
                                                    (float4*)xbuf, n4);
    }

    // ---- decoder ----
    gemm_kernel<64, false><<<gblocks, 256, 0, stream>>>(
        xbuf, nNodes, PK(8), dec1b, nullptr, abuf, 1,
        nullptr, nullptr, nullptr, nullptr, 0);
    decoder2_kernel<<<fblocks, 256, 0, stream>>>(abuf, dec2W, dec2b, feats,
                                                 (float*)d_out, nNodes);
}
